// Decoder_66005057405572
// MI455X (gfx1250) — matrix operand order repaired and checked by reading, not compiled, not run on hardware
//
#include <hip/hip_runtime.h>
#include <stdint.h>

// ---------------- types ----------------
typedef __attribute__((ext_vector_type(16))) __bf16 v16bf;
typedef __attribute__((ext_vector_type(8)))  float  v8f;
typedef __attribute__((ext_vector_type(4)))  unsigned int v4u;
typedef __attribute__((ext_vector_type(8)))  int    v8i;
typedef __attribute__((ext_vector_type(4)))  int    v4i;

#define B_    16
#define TIN   256
#define TOUT  150
#define E_    512
#define ARD_  1024
#define DRD_  1024
#define PRE_  256
#define AD_   128
#define NF_   32
#define KS_   31
#define NM_   80
#define PADC  15

#define NBLK  32
#define NTHR  256

#define WMMA_BF16(a,b,c) __builtin_amdgcn_wmma_f32_16x16x32_bf16(false,(a),false,(b),(short)0,(c),false,false)

#ifndef __has_builtin
#define __has_builtin(x) 0
#endif
#if __has_builtin(__builtin_amdgcn_tensor_load_to_lds)
#define HAVE_TDM 1
#else
#define HAVE_TDM 0
#endif

// ---------------- scalar helpers ----------------
__device__ inline __bf16 f2bf(float f){
  unsigned u = __builtin_bit_cast(unsigned, f);
  unsigned r = u + 0x7FFFu + ((u >> 16) & 1u);        // RNE
  unsigned short h = (unsigned short)(r >> 16);
  return __builtin_bit_cast(__bf16, h);
}
__device__ inline float bf2f(__bf16 b){
  unsigned short s = __builtin_bit_cast(unsigned short, b);
  unsigned u = ((unsigned)s) << 16;
  return __builtin_bit_cast(float, u);
}
__device__ inline float sigm(float x){ return 1.f/(1.f + __expf(-x)); }

// ---------------- WMMA fragment loads ----------------
// A tile (16 rows x 32 K, bf16, row-major source, ld in elems).
// Lane L: row = L&15, K chunks {h*8..h*8+7} and {16+h*8..16+h*8+7}, h = L>>4.
__device__ inline v16bf load_a_rm(const __bf16* base, int ld){
  int lane = threadIdx.x & 31;
  const __bf16* p = base + (size_t)(lane & 15) * ld + ((lane >> 4) * 8);
  v16bf a;
  *((uint4*)&a)     = *(const uint4*)p;
  *((uint4*)&a + 1) = *(const uint4*)(p + 16);
  return a;
}
// A tile from f32 source (converted on the fly).
__device__ inline v16bf load_a_f32(const float* base, int ld){
  int lane = threadIdx.x & 31;
  const float* p = base + (size_t)(lane & 15) * ld + ((lane >> 4) * 8);
  v16bf a;
#pragma unroll
  for (int j = 0; j < 8; ++j) a[j]     = f2bf(p[j]);
#pragma unroll
  for (int j = 0; j < 8; ++j) a[8 + j] = f2bf(p[16 + j]);
  return a;
}
// B tile: pre-packed 512 bf16 per tile, lane reads its 16 contiguous halves.
__device__ inline v16bf load_b_tile(const __bf16* wt){
  const __bf16* p = wt + (size_t)(threadIdx.x & 31) * 16;
  v16bf b;
  *((uint4*)&b)     = *(const uint4*)p;
  *((uint4*)&b + 1) = *(const uint4*)(p + 8);
  return b;
}
__device__ inline const __bf16* toff(const __bf16* w, int nt, int kt, int nKt){
  return w + ((size_t)nt * nKt + kt) * 512;
}

// ---------------- TDM: 1-D DMA of `bytes` (multiple of 8) global -> LDS -----
__device__ inline void tdm_issue(const void* gsrc, void* lds_dst, unsigned bytes){
#if HAVE_TDM
  unsigned long long ga = (unsigned long long)(uintptr_t)gsrc;
  unsigned lds_off = (unsigned)(uintptr_t)lds_dst;   // LDS aperture: low bits = offset
  unsigned n8 = bytes >> 3;                          // units of 8 bytes
  v4u g0;
  g0[0] = 1u;                                        // count=1, user descriptor
  g0[1] = lds_off;                                   // lds_addr
  g0[2] = (unsigned)(ga & 0xFFFFFFFFu);              // global_addr[31:0]
  g0[3] = (unsigned)((ga >> 32) & 0x01FFFFFFu) | (2u << 30); // addr[56:32] | type=2
  v8i g1;
  g1[0] = (int)(3u << 16);                           // wg_mask=0, data_size=3 (8B)
  g1[1] = (int)((n8 & 0xFFFFu) << 16);               // tensor_dim0[15:0]
  g1[2] = (int)(((n8 >> 16) & 0xFFFFu) | (1u << 16));// tensor_dim0[31:16], tensor_dim1=1
  g1[3] = (int)((n8 & 0xFFFFu) << 16);               // tile_dim0
  g1[4] = 0;                                         // tile_dim1=0, tile_dim2=0
  g1[5] = (int)n8;                                   // tensor_dim0_stride[31:0]
  g1[6] = 0;
  g1[7] = 0;
#if __clang_major__ >= 23
  v8i g4 = (v8i)0;
  __builtin_amdgcn_tensor_load_to_lds(g0, g1, (v4i)0, (v4i)0, g4, 0);
#else
  __builtin_amdgcn_tensor_load_to_lds(g0, g1, (v4i)0, (v4i)0, 0);
#endif
#endif
}

// stage h-state (32KB) into LDS: wave0 issues TDM (or all threads coop-copy)
__device__ inline void stage_issue(__bf16* lds_dst, const __bf16* src, int waveId){
#if HAVE_TDM
  if (waveId == 0) tdm_issue(src, lds_dst, (unsigned)(B_ * ARD_ * 2));
#else
  const uint4* s = (const uint4*)src;
  uint4*       d = (uint4*)lds_dst;
  for (int i = threadIdx.x; i < (B_ * ARD_ * 2) / 16; i += NTHR) d[i] = s[i];
#endif
}
__device__ inline void stage_wait(){
#if HAVE_TDM
#if __has_builtin(__builtin_amdgcn_s_wait_tensorcnt)
  __builtin_amdgcn_s_wait_tensorcnt(0);
#endif
#endif
  __syncthreads();
}

// ---------------- weight packing: W[N,K] f32 -> per-tile bf16 (B layout) ----
__global__ void pack_w(const float* __restrict__ W, __bf16* __restrict__ out,
                       int N, int K, int nKt){
  int tile = blockIdx.x;
  int nt = tile / nKt, kt = tile % nKt;
  int lane = threadIdx.x;
  int n  = nt * 16 + (lane & 15);
  int kb = kt * 32 + (lane >> 4) * 8;
  __bf16* o = out + (size_t)tile * 512 + (size_t)lane * 16;
#pragma unroll
  for (int j = 0; j < 16; ++j){
    int k = kb + (j < 8 ? j : j + 8);
    float v = (n < N && k < K) ? W[(size_t)n * K + k] : 0.f;
    o[j] = f2bf(v);
  }
}

// ---------------- pm = memory @ Wm^T  -> f32 [4096 x 128] -------------------
__global__ void k_pm(const float* __restrict__ mem, const __bf16* __restrict__ wm,
                     float* __restrict__ pm){
  int mt = blockIdx.x >> 3, nt = blockIdx.x & 7;
  v8f acc = {};
  const float* a0 = mem + (size_t)mt * 16 * E_;
#pragma unroll 2
  for (int kt = 0; kt < E_ / 32; ++kt){
    v16bf a = load_a_f32(a0 + kt * 32, E_);
    v16bf b = load_b_tile(toff(wm, nt, kt, E_ / 32));
    acc = WMMA_BF16(a, b, acc);
  }
  int lane = threadIdx.x;
  int col = nt * 16 + (lane & 15);
  int rb  = (lane >> 4) * 8;
#pragma unroll
  for (int g = 0; g < 8; ++g)
    pm[(size_t)(mt * 16 + rb + g) * AD_ + col] = acc[g];
}

// ---------------- prenet layer 1: seq[T,B,80] @ pW1^T, relu -> bf16 --------
__global__ void k_pre1(const float* __restrict__ dec, const __bf16* __restrict__ w1,
                       __bf16* __restrict__ s1){
  int mt = blockIdx.x >> 4, nt = blockIdx.x & 15;   // mt = timestep, rows = batch
  int lane = threadIdx.x;
  v8f acc = {};
  for (int kt = 0; kt < 3; ++kt){
    v16bf a;
    int bb = lane & 15;
#pragma unroll
    for (int j = 0; j < 16; ++j){
      int k = kt * 32 + (lane >> 4) * 8 + (j < 8 ? j : j + 8);
      float v = 0.f;
      if (k < NM_ && mt > 0) v = dec[(size_t)bb * NM_ * TOUT + (size_t)k * TOUT + (mt - 1)];
      a[j] = f2bf(v);
    }
    v16bf b = load_b_tile(toff(w1, nt, kt, 3));
    acc = WMMA_BF16(a, b, acc);
  }
  int col = nt * 16 + (lane & 15), rb = (lane >> 4) * 8;
#pragma unroll
  for (int g = 0; g < 8; ++g)
    s1[(size_t)(mt * 16 + rb + g) * PRE_ + col] = f2bf(fmaxf(acc[g], 0.f));
}

// ---------------- prenet layer 2 -------------------------------------------
__global__ void k_pre2(const __bf16* __restrict__ s1, const __bf16* __restrict__ w2,
                       __bf16* __restrict__ pre){
  int mt = blockIdx.x >> 4, nt = blockIdx.x & 15;
  v8f acc = {};
  const __bf16* a0 = s1 + (size_t)mt * 16 * PRE_;
#pragma unroll 2
  for (int kt = 0; kt < 8; ++kt){
    v16bf a = load_a_rm(a0 + kt * 32, PRE_);
    v16bf b = load_b_tile(toff(w2, nt, kt, 8));
    acc = WMMA_BF16(a, b, acc);
  }
  int lane = threadIdx.x;
  int col = nt * 16 + (lane & 15), rb = (lane >> 4) * 8;
#pragma unroll
  for (int g = 0; g < 8; ++g)
    pre[(size_t)(mt * 16 + rb + g) * PRE_ + col] = f2bf(fmaxf(acc[g], 0.f));
}

// ---------------- persistent decoder ---------------------------------------
struct P {
  const float  *memory, *loc_conv, *v;
  const float  *arn_bih, *arn_bhh, *drn_bih, *drn_bhh, *bp, *Wg, *bg;
  const __bf16 *wih_a, *whh_a, *wih_d, *whh_d, *wq, *wp, *ldense;
  const float  *pm;
  const __bf16 *pre;
  __bf16 *ah0, *ah1, *dh0, *dh1, *ctxb;
  float  *ctxf, *aw, *awc, *ebuf, *pq;
  float  *out_mel, *out_gate, *out_align;
  unsigned *sync;
};

__device__ void gsync(unsigned* s){
  __syncthreads();
  if (threadIdx.x == 0){
    __threadfence();
    unsigned gen = __hip_atomic_load(&s[1], __ATOMIC_ACQUIRE, __HIP_MEMORY_SCOPE_AGENT);
    if (atomicAdd(&s[0], 1u) == (unsigned)(NBLK - 1)){
      __hip_atomic_store(&s[0], 0u, __ATOMIC_RELAXED, __HIP_MEMORY_SCOPE_AGENT);
      __hip_atomic_fetch_add(&s[1], 1u, __ATOMIC_RELEASE, __HIP_MEMORY_SCOPE_AGENT);
    } else {
      while (__hip_atomic_load(&s[1], __ATOMIC_ACQUIRE, __HIP_MEMORY_SCOPE_AGENT) == gen)
        __builtin_amdgcn_s_sleep(1);
    }
  }
  __syncthreads();
}

__global__ __launch_bounds__(NTHR, 1) void decoder_persist(P p){
  const int bk = blockIdx.x, tid = threadIdx.x;
  const int w  = tid >> 5, lane = tid & 31;
  const int gw = bk * 8 + w;                        // global wave 0..255

  __shared__ float lds_g[8][256];                   // gate tiles (4 gates x 2 hidden tiles)
  __shared__ float lds_c[4][256];                   // persistent c: [0..1]=attn [2..3]=dec
  __shared__ __attribute__((aligned(16))) __bf16 lds_x[B_ * ARD_]; // staged h (32KB)
  for (int i = tid; i < 4 * 256; i += NTHR) ((float*)lds_c)[i] = 0.f;
  __syncthreads();

  for (int t = 0; t < TOUT; ++t){
    const __bf16* ahR = (t & 1) ? p.ah1 : p.ah0;    // h(t-1)
    __bf16*       ahW = (t & 1) ? p.ah0 : p.ah1;    // h(t)
    const __bf16* dhR = (t & 1) ? p.dh1 : p.dh0;
    __bf16*       dhW = (t & 1) ? p.dh0 : p.dh1;

    // ======== Phase 1: attention-LSTM gates: [x(pre_t), ctx] + h @ Whh ========
    stage_issue(lds_x, ahR, w);                     // TDM DMA of h(t-1) overlaps WMMAs
    {
      int ts = w >> 2, gate = w & 3;
      int ht = bk * 2 + ts;
      int nt = gate * 64 + ht;
      v8f acc = {};
      const __bf16* xpre = p.pre + (size_t)t * B_ * PRE_;
#pragma unroll 2
      for (int kt = 0; kt < 8; ++kt){
        v16bf a = load_a_rm(xpre + kt * 32, PRE_);
        acc = WMMA_BF16(a, load_b_tile(toff(p.wih_a, nt, kt, 24)), acc);
      }
#pragma unroll 2
      for (int kt = 0; kt < 16; ++kt){
        if (kt + 4 < 16) __builtin_prefetch(toff(p.wih_a, nt, 12 + kt, 24), 0, 1);
        v16bf a = load_a_rm(p.ctxb + kt * 32, E_);
        acc = WMMA_BF16(a, load_b_tile(toff(p.wih_a, nt, 8 + kt, 24)), acc);
      }
      stage_wait();                                 // s_wait_tensorcnt + barrier
#pragma unroll 4
      for (int kt = 0; kt < 32; ++kt){
        if (kt + 4 < 32) __builtin_prefetch(toff(p.whh_a, nt, kt + 4, 32), 0, 1);
        v16bf a = load_a_rm(lds_x + kt * 32, ARD_); // ds_load from staged h
        acc = WMMA_BF16(a, load_b_tile(toff(p.whh_a, nt, kt, 32)), acc);
      }
      int n = nt * 16 + (lane & 15);
      float bias = p.arn_bih[n] + p.arn_bhh[n];
      int rb = (lane >> 4) * 8;
#pragma unroll
      for (int g = 0; g < 8; ++g) lds_g[w][(rb + g) * 16 + (lane & 15)] = acc[g] + bias;
    }
    __syncthreads();
    for (int idx = tid; idx < 512; idx += NTHR){     // LSTM cell combine (i,f,g,o)
      int ts = idx >> 8, m = (idx >> 4) & 15, n = idx & 15, e = m * 16 + n;
      float gi = lds_g[ts*4+0][e], gf = lds_g[ts*4+1][e], gg = lds_g[ts*4+2][e], go = lds_g[ts*4+3][e];
      float c = sigm(gf) * lds_c[ts][e] + sigm(gi) * tanhf(gg);
      lds_c[ts][e] = c;
      int ht = bk * 2 + ts;
      ahW[(size_t)m * ARD_ + ht * 16 + n] = f2bf(sigm(go) * tanhf(c));
    }
    gsync(p.sync);                                   // S1: ah(t) visible

    // ======== pq = ah @ Wq^T (block 0 only) ========
    if (bk == 0){
      v8f acc = {};
#pragma unroll 2
      for (int kt = 0; kt < 32; ++kt){
        v16bf a = load_a_rm(ahW + kt * 32, ARD_);
        acc = WMMA_BF16(a, load_b_tile(toff(p.wq, w, kt, 32)), acc);
      }
      int col = w * 16 + (lane & 15), rb = (lane >> 4) * 8;
#pragma unroll
      for (int g = 0; g < 8; ++g) p.pq[(rb + g) * AD_ + col] = acc[g];
    }

    // ======== location conv -> A fragment (registers, all waves) ========
    v16bf aloc;
    {
      int r = lane & 15;
      int row = gw * 16 + r, b = row >> 8, tt = row & 255;
#pragma unroll
      for (int j = 0; j < 16; ++j){
        int f = (lane >> 4) * 8 + (j < 8 ? j : j + 8);
        float s = 0.f;
        for (int k = 0; k < KS_; ++k){
          int ti = tt + k - PADC;
          if (ti >= 0 && ti < TIN){
            s += p.aw [b * TIN + ti] * p.loc_conv[(f * 2 + 0) * KS_ + k]
               + p.awc[b * TIN + ti] * p.loc_conv[(f * 2 + 1) * KS_ + k];
          }
        }
        aloc[j] = f2bf(s);
      }
    }
    gsync(p.sync);                                   // S2a: pq visible

    // ======== pa (WMMA, K=32) fused with e = v . tanh(pq+pa+pm) ========
    {
      int r = lane & 15, rb = (lane >> 4) * 8;
      float s[8]; for (int g = 0; g < 8; ++g) s[g] = 0.f;
      for (int nt2 = 0; nt2 < 8; ++nt2){
        v8f acc = {};
        acc = WMMA_BF16(aloc, load_b_tile(toff(p.ldense, nt2, 0, 1)), acc);
        int n = nt2 * 16 + r;
        float vv = p.v[n];
#pragma unroll
        for (int g = 0; g < 8; ++g){
          int row = gw * 16 + rb + g, b = row >> 8;
          float x = acc[g] + p.pq[b * AD_ + n] + p.pm[(size_t)row * AD_ + n];
          s[g] += vv * tanhf(x);
        }
      }
#pragma unroll
      for (int g = 0; g < 8; ++g){
        float x = s[g];
        for (int m = 8; m >= 1; m >>= 1) x += __shfl_xor(x, m, 16);
        s[g] = x;
      }
      if (r == 0)
        for (int g = 0; g < 8; ++g) p.ebuf[gw * 16 + rb + g] = s[g];
    }
    gsync(p.sync);                                   // S2b: e visible

    // ======== softmax + alignments + cumulative weights (block 0) ========
    if (bk == 0){
      for (int rep = 0; rep < 2; ++rep){
        int b = w * 2 + rep;
        float vals[8], mx = -1e30f;
#pragma unroll
        for (int j = 0; j < 8; ++j){ vals[j] = p.ebuf[b * TIN + lane * 8 + j]; mx = fmaxf(mx, vals[j]); }
        for (int m = 16; m >= 1; m >>= 1) mx = fmaxf(mx, __shfl_xor(mx, m, 32));
        float sm = 0.f;
#pragma unroll
        for (int j = 0; j < 8; ++j){ vals[j] = __expf(vals[j] - mx); sm += vals[j]; }
        for (int m = 16; m >= 1; m >>= 1) sm += __shfl_xor(sm, m, 32);
        float inv = 1.f / sm;
#pragma unroll
        for (int j = 0; j < 8; ++j){
          int tt = lane * 8 + j;
          float wv = vals[j] * inv;
          p.aw [b * TIN + tt] = wv;
          p.awc[b * TIN + tt] += wv;
          p.out_align[(size_t)b * TOUT * TIN + (size_t)t * TIN + tt] = wv;
        }
      }
    }
    gsync(p.sync);                                   // S3: w visible

    // ======== ctx = w @ memory  (8192 threads = 16 x 512 outputs) ========
    {
      int gt = bk * NTHR + tid;
      int b = gt >> 9, col = gt & 511;
      const float* mrow = p.memory + (size_t)b * TIN * E_ + col;
      const float* wrow = p.aw + b * TIN;
      float s = 0.f;
      for (int tt = 0; tt < TIN; ++tt) s += wrow[tt] * mrow[(size_t)tt * E_];
      p.ctxf[b * E_ + col] = s;
      p.ctxb[b * E_ + col] = f2bf(s);
    }
    gsync(p.sync);                                   // S4: ctx visible

    // ======== Phase 3: decoder LSTM gates: [ah, ctx] + dh @ Whh ========
    stage_issue(lds_x, dhR, w);                      // TDM DMA of dh(t-1)
    {
      int ts = w >> 2, gate = w & 3;
      int ht = bk * 2 + ts;
      int nt = gate * 64 + ht;
      v8f acc = {};
#pragma unroll 2
      for (int kt = 0; kt < 32; ++kt){
        if (kt + 4 < 32) __builtin_prefetch(toff(p.wih_d, nt, kt + 4, 48), 0, 1);
        v16bf a = load_a_rm(ahW + kt * 32, ARD_);
        acc = WMMA_BF16(a, load_b_tile(toff(p.wih_d, nt, kt, 48)), acc);
      }
#pragma unroll 2
      for (int kt = 0; kt < 16; ++kt){
        v16bf a = load_a_rm(p.ctxb + kt * 32, E_);
        acc = WMMA_BF16(a, load_b_tile(toff(p.wih_d, nt, 32 + kt, 48)), acc);
      }
      stage_wait();                                  // s_wait_tensorcnt + barrier
#pragma unroll 4
      for (int kt = 0; kt < 32; ++kt){
        if (kt + 4 < 32) __builtin_prefetch(toff(p.whh_d, nt, kt + 4, 32), 0, 1);
        v16bf a = load_a_rm(lds_x + kt * 32, DRD_);  // ds_load from staged dh
        acc = WMMA_BF16(a, load_b_tile(toff(p.whh_d, nt, kt, 32)), acc);
      }
      int n = nt * 16 + (lane & 15);
      float bias = p.drn_bih[n] + p.drn_bhh[n];
      int rb = (lane >> 4) * 8;
#pragma unroll
      for (int g = 0; g < 8; ++g) lds_g[w][(rb + g) * 16 + (lane & 15)] = acc[g] + bias;
    }
    __syncthreads();
    for (int idx = tid; idx < 512; idx += NTHR){
      int ts = idx >> 8, m = (idx >> 4) & 15, n = idx & 15, e = m * 16 + n;
      float gi = lds_g[ts*4+0][e], gf = lds_g[ts*4+1][e], gg = lds_g[ts*4+2][e], go = lds_g[ts*4+3][e];
      float c = sigm(gf) * lds_c[2 + ts][e] + sigm(gi) * tanhf(gg);
      lds_c[2 + ts][e] = c;
      int ht = bk * 2 + ts;
      dhW[(size_t)m * DRD_ + ht * 16 + n] = f2bf(sigm(go) * tanhf(c));
    }
    gsync(p.sync);                                   // S5: dh visible

    // ======== Phase 4: mel / gate outputs (block 1) ========
    if (bk == 1 && w < 5){
      v8f acc = {};
#pragma unroll 2
      for (int kt = 0; kt < 32; ++kt){
        v16bf a = load_a_rm(dhW + kt * 32, DRD_);
        acc = WMMA_BF16(a, load_b_tile(toff(p.wp, w, kt, 48)), acc);
      }
#pragma unroll 2
      for (int kt = 0; kt < 16; ++kt){
        v16bf a = load_a_rm(p.ctxb + kt * 32, E_);
        acc = WMMA_BF16(a, load_b_tile(toff(p.wp, w, 32 + kt, 48)), acc);
      }
      int n = w * 16 + (lane & 15), rb = (lane >> 4) * 8;
      float bias = p.bp[n];
#pragma unroll
      for (int g = 0; g < 8; ++g){
        int m = rb + g;
        p.out_mel[(size_t)m * NM_ * TOUT + (size_t)n * TOUT + t] = acc[g] + bias;
      }
    }
    if (bk == 1 && w == 5){
      int b = lane >> 1, half = lane & 1;
      float s = 0.f;
      for (int k = half * 768; k < half * 768 + 768; ++k){
        float hv = (k < DRD_) ? bf2f(dhW[(size_t)b * DRD_ + k]) : p.ctxf[b * E_ + (k - DRD_)];
        s += hv * p.Wg[k];
      }
      s += __shfl_xor(s, 1, 32);
      if (half == 0) p.out_gate[b * TOUT + t] = s + p.bg[0];
    }
    // next iteration's Phase-1 reads only {pre, ctxb, ahR'} — all stable here.
  }
}

// ---------------- host ----------------
extern "C" void kernel_launch(void* const* d_in, const int* in_sizes, int n_in,
                              void* d_out, int out_size, void* d_ws, size_t ws_size,
                              hipStream_t stream){
  (void)in_sizes; (void)n_in; (void)out_size; (void)ws_size;
  const float* memory   = (const float*)d_in[0];
  const float* dec      = (const float*)d_in[1];
  const float* pW1      = (const float*)d_in[2];
  const float* pW2      = (const float*)d_in[3];
  const float* arn_Wih  = (const float*)d_in[4];
  const float* arn_Whh  = (const float*)d_in[5];
  const float* arn_bih  = (const float*)d_in[6];
  const float* arn_bhh  = (const float*)d_in[7];
  const float* Wq       = (const float*)d_in[8];
  const float* Wm       = (const float*)d_in[9];
  const float* v        = (const float*)d_in[10];
  const float* loc_conv = (const float*)d_in[11];
  const float* ldense   = (const float*)d_in[12];
  const float* drn_Wih  = (const float*)d_in[13];
  const float* drn_Whh  = (const float*)d_in[14];
  const float* drn_bih  = (const float*)d_in[15];
  const float* drn_bhh  = (const float*)d_in[16];
  const float* Wp       = (const float*)d_in[17];
  const float* bp       = (const float*)d_in[18];
  const float* Wg       = (const float*)d_in[19];
  const float* bg       = (const float*)d_in[20];

  char* base = (char*)d_ws; size_t off = 0;
  auto alloc = [&](size_t bytes)->void*{
    void* p = base + off; off = (off + bytes + 255) & ~(size_t)255; return p;
  };

  // --- zero-initialized per-call state (memset below) ---
  unsigned* syncp = (unsigned*)alloc(256);
  __bf16* ah0  = (__bf16*)alloc((size_t)B_ * ARD_ * 2);
  __bf16* ah1  = (__bf16*)alloc((size_t)B_ * ARD_ * 2);
  __bf16* dh0  = (__bf16*)alloc((size_t)B_ * DRD_ * 2);
  __bf16* dh1  = (__bf16*)alloc((size_t)B_ * DRD_ * 2);
  __bf16* ctxb = (__bf16*)alloc((size_t)B_ * E_ * 2);
  float*  ctxf = (float*)alloc((size_t)B_ * E_ * 4);
  float*  aw   = (float*)alloc((size_t)B_ * TIN * 4);
  float*  awc  = (float*)alloc((size_t)B_ * TIN * 4);
  float*  ebuf = (float*)alloc((size_t)B_ * TIN * 4);
  float*  pq   = (float*)alloc((size_t)B_ * AD_ * 4);
  size_t stateBytes = off;
  hipMemsetAsync(d_ws, 0, stateBytes, stream);

  // --- packed weights (bf16, tiled) + precomputed tensors ---
  __bf16* wih_a = (__bf16*)alloc((size_t)4096 * 768  * 2);
  __bf16* whh_a = (__bf16*)alloc((size_t)4096 * 1024 * 2);
  __bf16* wih_d = (__bf16*)alloc((size_t)4096 * 1536 * 2);
  __bf16* whh_d = (__bf16*)alloc((size_t)4096 * 1024 * 2);
  __bf16* wq_p  = (__bf16*)alloc((size_t)128 * 1024 * 2);
  __bf16* wm_p  = (__bf16*)alloc((size_t)128 * 512  * 2);
  __bf16* wp_p  = (__bf16*)alloc((size_t)5 * 48 * 512 * 2);
  __bf16* ld_p  = (__bf16*)alloc((size_t)8 * 1 * 512 * 2);
  __bf16* w1_p  = (__bf16*)alloc((size_t)16 * 3 * 512 * 2);
  __bf16* w2_p  = (__bf16*)alloc((size_t)16 * 8 * 512 * 2);
  float*  pm    = (float*)alloc((size_t)B_ * TIN * AD_ * 4);
  __bf16* s1    = (__bf16*)alloc((size_t)TOUT * B_ * PRE_ * 2);
  __bf16* pre   = (__bf16*)alloc((size_t)TOUT * B_ * PRE_ * 2);

  pack_w<<<256 * 24, 32, 0, stream>>>(arn_Wih, wih_a, 4096,  768, 24);
  pack_w<<<256 * 32, 32, 0, stream>>>(arn_Whh, whh_a, 4096, 1024, 32);
  pack_w<<<256 * 48, 32, 0, stream>>>(drn_Wih, wih_d, 4096, 1536, 48);
  pack_w<<<256 * 32, 32, 0, stream>>>(drn_Whh, whh_d, 4096, 1024, 32);
  pack_w<<<  8 * 32, 32, 0, stream>>>(Wq,      wq_p,   128, 1024, 32);
  pack_w<<<  8 * 16, 32, 0, stream>>>(Wm,      wm_p,   128,  512, 16);
  pack_w<<<  5 * 48, 32, 0, stream>>>(Wp,      wp_p,    80, 1536, 48);
  pack_w<<<  8 *  1, 32, 0, stream>>>(ldense,  ld_p,   128,   32,  1);
  pack_w<<< 16 *  3, 32, 0, stream>>>(pW1,     w1_p,   256,   80,  3);
  pack_w<<< 16 *  8, 32, 0, stream>>>(pW2,     w2_p,   256,  256,  8);

  k_pm  <<<256 * 8, 32, 0, stream>>>(memory, wm_p, pm);
  k_pre1<<<TOUT * 16, 32, 0, stream>>>(dec, w1_p, s1);
  k_pre2<<<TOUT * 16, 32, 0, stream>>>(s1, w2_p, pre);

  P p;
  p.memory = memory; p.loc_conv = loc_conv; p.v = v;
  p.arn_bih = arn_bih; p.arn_bhh = arn_bhh; p.drn_bih = drn_bih; p.drn_bhh = drn_bhh;
  p.bp = bp; p.Wg = Wg; p.bg = bg;
  p.wih_a = wih_a; p.whh_a = whh_a; p.wih_d = wih_d; p.whh_d = whh_d;
  p.wq = wq_p; p.wp = wp_p; p.ldense = ld_p;
  p.pm = pm; p.pre = pre;
  p.ah0 = ah0; p.ah1 = ah1; p.dh0 = dh0; p.dh1 = dh1; p.ctxb = ctxb;
  p.ctxf = ctxf; p.aw = aw; p.awc = awc; p.ebuf = ebuf; p.pq = pq;
  float* out = (float*)d_out;
  p.out_mel = out;                            // [16, 80, 150]
  p.out_gate = out + (size_t)B_ * NM_ * TOUT; // [16, 150]
  p.out_align = p.out_gate + (size_t)B_ * TOUT; // [16, 150, 256]
  p.sync = syncp;

  decoder_persist<<<NBLK, NTHR, 0, stream>>>(p);
}
